// Transformer_1116691497028
// MI455X (gfx1250) — matrix operand order repaired and checked by reading, not compiled, not run on hardware
//
#include <hip/hip_runtime.h>

typedef unsigned short u16;
typedef __attribute__((ext_vector_type(16))) __bf16 v16bf;
typedef __attribute__((ext_vector_type(8)))  float  v8f;

struct U8 { unsigned u[8]; };

static __device__ __forceinline__ u16 f2b(float f) {
  unsigned u = __builtin_bit_cast(unsigned, f);
  u += 0x7FFFu + ((u >> 16) & 1u);  // round-to-nearest-even
  return (u16)(u >> 16);
}

static __device__ __forceinline__ v8f wmma_bf16(v16bf a, v16bf b, v8f c) {
  return __builtin_amdgcn_wmma_f32_16x16x32_bf16(false, a, false, b, (short)0, c,
                                                 false, false);
}

// ---- CDNA5 async global->LDS copy (ASYNCcnt) ------------------------------
static __device__ __forceinline__ void async_b128(unsigned lds, const void* g) {
  asm volatile("global_load_async_to_lds_b128 %0, %1, off"
               :: "v"(lds), "v"(g) : "memory");
}
static __device__ __forceinline__ void wait_async() {
  asm volatile("s_wait_asynccnt 0" ::: "memory");
}
static __device__ __forceinline__ unsigned lds_addr(const void* p) {
  return (unsigned)(uintptr_t)p;  // generic LDS ptr: addr[31:0] = LDS offset
}

// A-fragment (16x32 bf16): lane holds row (mrow+lm); K pairs across 8 VGPRs.
static __device__ __forceinline__ v16bf load_afrag(const u16* src, int mrow,
                                                   int ld, int kbase) {
  int lane = threadIdx.x & 31, half = lane >> 4, lm = lane & 15;
  U8 t;
#pragma unroll
  for (int v = 0; v < 8; ++v) {
    int k = kbase + (v >> 2) * 16 + half * 8 + (v & 3) * 2;
    t.u[v] = *(const unsigned*)(src + (mrow + lm) * ld + k);
  }
  return __builtin_bit_cast(v16bf, t);
}

// B-fragment (32x16 bf16): lane holds col (ncol+lm); src stored [n][k].
static __device__ __forceinline__ v16bf load_bfrag(const u16* src, int ncol,
                                                   int ld, int kbase) {
  int lane = threadIdx.x & 31, half = lane >> 4, lm = lane & 15;
  U8 t;
#pragma unroll
  for (int v = 0; v < 8; ++v) {
    int k = kbase + (v >> 2) * 16 + half * 8 + (v & 3) * 2;
    t.u[v] = *(const unsigned*)(src + (ncol + lm) * ld + k);
  }
  return __builtin_bit_cast(v16bf, t);
}

// ---------------------------------------------------------------- converts
__global__ void cvt_f32_bf16(const float* __restrict__ s, u16* __restrict__ d,
                             int n) {
  int i = blockIdx.x * 256 + threadIdx.x;
  if (i < n) d[i] = f2b(s[i]);
}

// Transpose-convert: in fp32 [K][N] -> out bf16 [N][K]. 32x32 LDS tiles.
__global__ void tconv(const float* __restrict__ in, u16* __restrict__ out,
                      int K, int N) {
  __shared__ float tile[32][33];
  int bx = blockIdx.x, by = blockIdx.y;
  int t = threadIdx.x, tx = t & 31, ty0 = t >> 5;
#pragma unroll
  for (int j = 0; j < 4; ++j) {
    int ty = ty0 + j * 8;
    tile[ty][tx] = in[(size_t)(by * 32 + ty) * N + bx * 32 + tx];
  }
  __syncthreads();
#pragma unroll
  for (int j = 0; j < 4; ++j) {
    int ty = ty0 + j * 8;
    out[(size_t)(bx * 32 + ty) * K + by * 32 + tx] = f2b(tile[tx][ty]);
  }
}

// Wq[i] (H=16, D=1024, DK=64) fp32 -> bf16 [(h,e)][d] = [1024][1024]
__global__ void perm_qkv_t(const float* __restrict__ in, u16* __restrict__ out) {
  __shared__ float tile[32][33];
  int bx = blockIdx.x, by = blockIdx.y, h = blockIdx.z;
  int t = threadIdx.x, tx = t & 31, ty0 = t >> 5;
#pragma unroll
  for (int j = 0; j < 4; ++j) {
    int ty = ty0 + j * 8;
    tile[ty][tx] = in[((size_t)h * 1024 + by * 32 + ty) * 64 + bx * 32 + tx];
  }
  __syncthreads();
#pragma unroll
  for (int j = 0; j < 4; ++j) {
    int ty = ty0 + j * 8;
    out[((size_t)h * 64 + bx * 32 + ty) * 1024 + by * 32 + tx] = f2b(tile[tx][ty]);
  }
}

// ---------------------------------------------------------------- GEMM
// C[M,N] = epilogue( A[M,K]_bf16 * Wt[N,K]_bf16 )   (weights pre-transposed)
// block tile 128x256, 8 waves (2x4), wave tile 64x64 (16 wmma / K-step),
// K step 32, double-buffered LDS, async b128 staging for both tiles.
#define LDT 40  // 80B rows: 16B-aligned, 20-bank stride -> conflict free

static __device__ __forceinline__ void gemm_stage(const u16* __restrict__ A,
                                                  const u16* __restrict__ Wt,
                                                  u16* as, u16* bst, int m0,
                                                  int n0, int kt, int K) {
  int t = threadIdx.x;
#pragma unroll
  for (int i = 0; i < 2; ++i) {  // A: 128 rows x 4 chunks(16B)
    int idx = i * 256 + t, r = idx >> 2, c = idx & 3;
    async_b128(lds_addr(as + r * LDT) + c * 16,
               (const char*)A + ((size_t)(m0 + r) * K + kt * 32) * 2 + c * 16);
  }
#pragma unroll
  for (int i = 0; i < 4; ++i) {  // B: 256 n-rows x 4 chunks(16B)
    int idx = i * 256 + t, r = idx >> 2, c = idx & 3;
    async_b128(lds_addr(bst + r * LDT) + c * 16,
               (const char*)Wt + ((size_t)(n0 + r) * K + kt * 32) * 2 + c * 16);
  }
}

__global__ __launch_bounds__(256) void gemm_bf16(
    const u16* __restrict__ A, const u16* __restrict__ Wt,
    const float* __restrict__ bias, const float* __restrict__ resid,
    float* __restrict__ outf, u16* __restrict__ outb, int M, int N, int K,
    int mode)  // mode: 0 plain, 1 gelu, 2 (out+res)*0.5
{
  __shared__ __attribute__((aligned(16))) u16 As[2][128 * LDT];
  __shared__ __attribute__((aligned(16))) u16 Bs[2][256 * LDT];
  int m0 = blockIdx.y * 128, n0 = blockIdx.x * 256;
  int t = threadIdx.x, w = t >> 5, lane = t & 31;
  int half = lane >> 4, lm = lane & 15;
  int wr = w >> 2, wc = w & 3;  // 2x4 wave grid, wave tile 64x64

  v8f acc[4][4];
#pragma unroll
  for (int ms = 0; ms < 4; ++ms)
#pragma unroll
    for (int ns = 0; ns < 4; ++ns) acc[ms][ns] = (v8f){};

  int KT = K / 32, buf = 0;
  gemm_stage(A, Wt, As[0], Bs[0], m0, n0, 0, K);
  for (int kt = 0; kt < KT; ++kt) {
    wait_async();     // tile(kt) landed in LDS (own wave's copies)
    __syncthreads();  // all waves' tiles visible
    if (kt + 1 < KT) gemm_stage(A, Wt, As[buf ^ 1], Bs[buf ^ 1], m0, n0, kt + 1, K);
    v16bf a[4];
#pragma unroll
    for (int ms = 0; ms < 4; ++ms)
      a[ms] = load_afrag(As[buf], wr * 64 + ms * 16, LDT, 0);
#pragma unroll
    for (int ns = 0; ns < 4; ++ns) {
      v16bf bb = load_bfrag(Bs[buf], wc * 64 + ns * 16, LDT, 0);
#pragma unroll
      for (int ms = 0; ms < 4; ++ms)
        acc[ms][ns] = wmma_bf16(a[ms], bb, acc[ms][ns]);
    }
    __syncthreads();  // done reading buf before it is re-staged
    buf ^= 1;
  }

  // epilogue: C layout -> row = half*8 + r (per VGPR r), col = lm
#pragma unroll
  for (int ms = 0; ms < 4; ++ms)
#pragma unroll
    for (int ns = 0; ns < 4; ++ns) {
      int colb = n0 + wc * 64 + ns * 16 + lm;
      float bv = bias ? bias[colb] : 0.0f;
#pragma unroll
      for (int r = 0; r < 8; ++r) {
        int row = m0 + wr * 64 + ms * 16 + half * 8 + r;
        float v = acc[ms][ns][r] + bv;
        if (mode == 1) v = 0.5f * v * (1.0f + erff(v * 0.70710678118654752f));
        if (mode == 2) v = (v + resid[(size_t)row * N + colb]) * 0.5f;
        if (outf) outf[(size_t)row * N + colb] = v;
        if (outb) outb[(size_t)row * N + colb] = f2b(v);
      }
    }
}

// ---------------------------------------------------------------- flash attn
// grid (L/128, B*H). block = 256 = 8 waves; wave owns 16 query rows.
// q,k,v live in the fused [B*L][3072] buffer (row stride QS, head offset h*64);
// o is bf16 [B*L][1024]. No 1/sqrt(dk) scale (matches reference).
#define LDF 72  // 144B rows: 16B-aligned, 36-bank stride -> conflict free
#define QS 3072

__global__ __launch_bounds__(256) void flash_attn(
    const u16* __restrict__ qg, const u16* __restrict__ kg,
    const u16* __restrict__ vg, const int* __restrict__ maskg,
    u16* __restrict__ og) {
  const int L = 1024, DM = 1024;
  int qb = blockIdx.x, b = blockIdx.y >> 4, h = blockIdx.y & 15;
  int t = threadIdx.x, w = t >> 5, lane = t & 31;
  int half = lane >> 4, lm = lane & 15;

  __shared__ __attribute__((aligned(16))) u16 Qs[128 * LDF];  // [qrow][d]
  __shared__ __attribute__((aligned(16))) u16 Ks[64 * LDF];   // [key][d]
  __shared__ __attribute__((aligned(16))) u16 Vst[64 * LDF];  // [e][key]
  __shared__ __attribute__((aligned(16))) u16 Ps[8 * 16 * LDF];

  const size_t qbase = ((size_t)b * L + qb * 128) * QS + h * 64;
#pragma unroll
  for (int i = 0; i < 4; ++i) {  // Q: 128 rows x 8 chunks(16B), async
    int idx = i * 256 + t, r = idx >> 3, c = idx & 7;
    async_b128(lds_addr(Qs + r * LDF) + c * 16,
               (const char*)qg + (qbase + (size_t)r * QS) * 2 + c * 16);
  }

  v8f oacc[4];
  float mrow[8], lrow[8];
  int q0 = qb * 128 + w * 16;
  int mrm[8];
#pragma unroll
  for (int r = 0; r < 8; ++r) {
    mrow[r] = -3e38f;
    lrow[r] = 0.0f;
    mrm[r] = maskg[b * L + q0 + half * 8 + r];
  }
#pragma unroll
  for (int nt = 0; nt < 4; ++nt) oacc[nt] = (v8f){};

  for (int jb = 0; jb < 16; ++jb) {
    __syncthreads();  // prev iteration's reads of Ks/Vst complete
    size_t kvb = ((size_t)b * L + jb * 64) * QS + h * 64;
#pragma unroll
    for (int i = 0; i < 2; ++i) {  // K: 64 rows x 8 chunks(16B), async
      int idx = i * 256 + t, r = idx >> 3, c = idx & 7;
      async_b128(lds_addr(Ks + r * LDF) + c * 16,
                 (const char*)kg + (kvb + (size_t)r * QS) * 2 + c * 16);
    }
    // V transposed: two-phase register staging (loads batched, then stores)
    u16 tv[16];
#pragma unroll
    for (int i = 0; i < 16; ++i) {
      int idx = i * 256 + t, j = idx >> 6, e = idx & 63;
      tv[i] = vg[kvb + (size_t)j * QS + e];
    }
#pragma unroll
    for (int i = 0; i < 16; ++i) {
      int idx = i * 256 + t, j = idx >> 6, e = idx & 63;
      Vst[e * LDF + j] = tv[i];
    }
    wait_async();     // Q (first iter) + K copies landed
    __syncthreads();  // publish to all waves

    // S = Q * K^T  (16x64 per wave)
    v16bf aq0 = load_afrag(Qs, w * 16, LDF, 0);
    v16bf aq1 = load_afrag(Qs, w * 16, LDF, 32);
    v8f s[4];
#pragma unroll
    for (int nt = 0; nt < 4; ++nt) {
      s[nt] = (v8f){};
      v16bf b0 = load_bfrag(Ks, nt * 16, LDF, 0);
      s[nt] = wmma_bf16(aq0, b0, s[nt]);
      v16bf b1 = load_bfrag(Ks, nt * 16, LDF, 32);
      s[nt] = wmma_bf16(aq1, b1, s[nt]);
    }

    int mcol[4];
#pragma unroll
    for (int nt = 0; nt < 4; ++nt)
      mcol[nt] = maskg[b * L + jb * 64 + nt * 16 + lm];

    // online softmax per row (C-layout: lane=col, rows split by lane-half)
#pragma unroll
    for (int r = 0; r < 8; ++r) {
#pragma unroll
      for (int nt = 0; nt < 4; ++nt)
        if (!(mrm[r] && mcol[nt])) s[nt][r] = -1e30f;
      float rm = fmaxf(fmaxf(s[0][r], s[1][r]), fmaxf(s[2][r], s[3][r]));
#pragma unroll
      for (int mk = 1; mk <= 8; mk <<= 1) rm = fmaxf(rm, __shfl_xor(rm, mk));
      float mn = fmaxf(mrow[r], rm);
      float sc = __expf(mrow[r] - mn);
      float rs = 0.0f;
#pragma unroll
      for (int nt = 0; nt < 4; ++nt) {
        float p = __expf(s[nt][r] - mn);
        s[nt][r] = p;
        rs += p;
      }
#pragma unroll
      for (int mk = 1; mk <= 8; mk <<= 1) rs += __shfl_xor(rs, mk);
      lrow[r] = lrow[r] * sc + rs;
      mrow[r] = mn;
#pragma unroll
      for (int nt = 0; nt < 4; ++nt) {
        oacc[nt][r] *= sc;
        Ps[(w * 16 + half * 8 + r) * LDF + nt * 16 + lm] = f2b(s[nt][r]);
      }
    }
    // DS ops are in-order within a wave; wait before re-reading wave-private P
    asm volatile("s_wait_dscnt 0" ::: "memory");

    // O += P * V
    v16bf ap0 = load_afrag(Ps, w * 16, LDF, 0);
    v16bf ap1 = load_afrag(Ps, w * 16, LDF, 32);
#pragma unroll
    for (int nt = 0; nt < 4; ++nt) {
      v16bf b0 = load_bfrag(Vst, nt * 16, LDF, 0);
      oacc[nt] = wmma_bf16(ap0, b0, oacc[nt]);
      v16bf b1 = load_bfrag(Vst, nt * 16, LDF, 32);
      oacc[nt] = wmma_bf16(ap1, b1, oacc[nt]);
    }
  }

#pragma unroll
  for (int r = 0; r < 8; ++r) {
    float inv = 1.0f / (lrow[r] + 1e-20f);
    int row = q0 + half * 8 + r;
#pragma unroll
    for (int nt = 0; nt < 4; ++nt)
      og[((size_t)b * L + row) * DM + h * 64 + nt * 16 + lm] =
          f2b(oacc[nt][r] * inv);
  }
}

// ---------------------------------------------------------------- launcher
extern "C" void kernel_launch(void* const* d_in, const int* in_sizes, int n_in,
                              void* d_out, int out_size, void* d_ws,
                              size_t ws_size, hipStream_t stream) {
  const int B = 4, L = 1024, D = 1024, DFF = 2048, M = B * L;  // 4096
  const float* x    = (const float*)d_in[0];
  const int*   mask = (const int*)d_in[1];
  const float* Wq   = (const float*)d_in[2];
  const float* Wk   = (const float*)d_in[3];
  const float* Wv   = (const float*)d_in[4];
  const float* Wo   = (const float*)d_in[5];
  const float* W1   = (const float*)d_in[6];
  const float* b1   = (const float*)d_in[7];
  const float* W2   = (const float*)d_in[8];
  const float* b2   = (const float*)d_in[9];

  char* p = (char*)d_ws;
  auto carve = [&](size_t bytes) {
    void* r = (void*)p;
    p += (bytes + 255) & ~size_t(255);
    return r;
  };
  u16* xb   = (u16*)carve((size_t)M * D * 2);
  u16* qkvb = (u16*)carve((size_t)M * 3 * D * 2);  // [M][3072]: q|k|v
  u16* ob   = (u16*)carve((size_t)M * D * 2);
  u16* h1   = (u16*)carve((size_t)M * DFF * 2);
  float* xf = (float*)carve((size_t)M * D * 4);
  u16* wqkv = (u16*)carve((size_t)3 * D * D * 2);  // [3072][1024]
  u16* wob  = (u16*)carve((size_t)D * D * 2);      // [1024][1024]
  u16* w1b  = (u16*)carve((size_t)D * DFF * 2);    // [2048][1024]
  u16* w2b  = (u16*)carve((size_t)DFF * D * 2);    // [1024][2048]

  cvt_f32_bf16<<<(M * D) / 256, 256, 0, stream>>>(x, xb, M * D);

  for (int i = 0; i < 6; ++i) {
    perm_qkv_t<<<dim3(2, 32, 16), 256, 0, stream>>>(Wq + (size_t)i * D * D, wqkv);
    perm_qkv_t<<<dim3(2, 32, 16), 256, 0, stream>>>(Wk + (size_t)i * D * D,
                                                    wqkv + (size_t)D * D);
    perm_qkv_t<<<dim3(2, 32, 16), 256, 0, stream>>>(Wv + (size_t)i * D * D,
                                                    wqkv + (size_t)2 * D * D);
    tconv<<<dim3(D / 32, D / 32), 256, 0, stream>>>(Wo + (size_t)i * D * D, wob, D, D);
    tconv<<<dim3(DFF / 32, D / 32), 256, 0, stream>>>(W1 + (size_t)i * D * DFF, w1b, D, DFF);
    tconv<<<dim3(D / 32, DFF / 32), 256, 0, stream>>>(W2 + (size_t)i * DFF * D, w2b, DFF, D);

    // fused Q|K|V projection: [M,1024] x [1024,3072] -> [M,3072] bf16
    gemm_bf16<<<dim3((3 * D) / 256, M / 128), 256, 0, stream>>>(
        xb, wqkv, nullptr, nullptr, nullptr, qkvb, M, 3 * D, D, 0);

    flash_attn<<<dim3(L / 128, B * 16), 256, 0, stream>>>(
        qkvb, qkvb + D, qkvb + 2 * D, mask, ob);

    // x = (o @ Wo + x) * 0.5   -> fp32 x and bf16 xb
    const float* resid0 = (i == 0) ? x : xf;
    gemm_bf16<<<dim3(D / 256, M / 128), 256, 0, stream>>>(
        ob, wob, nullptr, resid0, xf, xb, M, D, D, 2);

    // f = gelu(x @ W1 + b1) -> bf16
    gemm_bf16<<<dim3(DFF / 256, M / 128), 256, 0, stream>>>(
        xb, w1b, b1 + (size_t)i * DFF, nullptr, nullptr, h1, M, DFF, D, 1);

    // x = (f @ W2 + b2 + x) * 0.5 ; last layer writes d_out directly
    float* outx = (i == 5) ? (float*)d_out : xf;
    gemm_bf16<<<dim3(D / 256, M / 128), 256, 0, stream>>>(
        h1, w2b, b2 + (size_t)i * D, xf, outx, xb, M, D, DFF, 2);
  }
  (void)in_sizes; (void)n_in; (void)out_size; (void)ws_size;
}
